// SRNO_26310969655722
// MI455X (gfx1250) — matrix operand order repaired and checked by reading, not compiled, not run on hardware
//
#include <hip/hip_runtime.h>
#include <math.h>

// ---------------------------------------------------------------------------
// SRNO super-resolution pipeline for MI455X (gfx1250, wave32, WMMA).
// Heavy lifting: f16 WMMA GEMMs (v_wmma_f32_16x16x32_f16, f32 accum) with
// fused bias / GELU / residual epilogues and a software-pipelined
// (register double-buffered) K loop. Everything else is cheap VALU work.
// ---------------------------------------------------------------------------

typedef __attribute__((ext_vector_type(16))) _Float16 v16h;
typedef __attribute__((ext_vector_type(8)))  float    v8f;
typedef __attribute__((ext_vector_type(4)))  float    v4f;

#define HLR 80
#define WLR 80
#define NPIX_LR (HLR * WLR)        // 6400
#define HHR 320
#define WHR 320
#define MQ (HHR * WHR)             // 102400 query pixels
#define WIDTH 256
#define HEADS 16
#define KP0 544                    // 522 grid channels padded to K%32==0
#define K522 522

union HFrag { v16h h; v4f q[2]; };

// ---------------------------------------------------------------------------
// Encoder: generic 3x3 SAME conv on 80x80, optional ReLU and residual add.
// ~4.3 GFLOP total -> negligible; plain VALU kernel.
// ---------------------------------------------------------------------------
__global__ void k_conv3x3(const float* __restrict__ in, int Cin,
                          const float* __restrict__ w, const float* __restrict__ b,
                          float* __restrict__ out, int Cout,
                          const float* __restrict__ addsrc, int relu)
{
    int idx = blockIdx.x * blockDim.x + threadIdx.x;
    int total = Cout * NPIX_LR;
    if (idx >= total) return;
    int co = idx / NPIX_LR;
    int p  = idx % NPIX_LR;
    int y = p / WLR, x = p % WLR;
    float acc = b[co];
    const float* wc = w + (size_t)co * Cin * 9;
    for (int ci = 0; ci < Cin; ++ci) {
        const float* ip = in + (size_t)ci * NPIX_LR;
        const float* wk = wc + ci * 9;
        #pragma unroll
        for (int ky = 0; ky < 3; ++ky) {
            int yy = y + ky - 1;
            if (yy < 0 || yy >= HLR) continue;
            #pragma unroll
            for (int kx = 0; kx < 3; ++kx) {
                int xx = x + kx - 1;
                if (xx < 0 || xx >= WLR) continue;
                acc += ip[yy * WLR + xx] * wk[ky * 3 + kx];
            }
        }
    }
    if (relu) acc = fmaxf(acc, 0.f);
    if (addsrc) acc += addsrc[idx];
    out[idx] = acc;
}

// ---------------------------------------------------------------------------
// Build the 522-channel (padded to 544) grid row per query pixel, as f16 A
// matrix [MQ, 544]: 4 corner gathers + area weights + sin modulation.
// ---------------------------------------------------------------------------
__global__ void k_build_grid(const float* __restrict__ feat,
                             const float* __restrict__ coord,
                             const float* __restrict__ cell,
                             const float* __restrict__ modw,
                             const float* __restrict__ modb,
                             _Float16* __restrict__ grid)
{
    int q = blockIdx.x * blockDim.x + threadIdx.x;
    if (q >= MQ) return;
    float cy = coord[2 * q + 0], cx = coord[2 * q + 1];
    const float rx = 1.f / HLR, ry = 1.f / WLR, eps = 1e-6f;
    float rel0[4], rel1[4], area[4];
    int pix[4];
    int j = 0;
    for (int vx = -1; vx <= 1; vx += 2) {
        for (int vy = -1; vy <= 1; vy += 2, ++j) {
            float cyc = fminf(fmaxf(cy + vx * rx + eps, -1.f + 1e-6f), 1.f - 1e-6f);
            float cxc = fminf(fmaxf(cx + vy * ry + eps, -1.f + 1e-6f), 1.f - 1e-6f);
            int iy = min(max((int)floorf(((cyc + 1.f) * HLR - 1.f) * 0.5f + 0.5f), 0), HLR - 1);
            int ix = min(max((int)floorf(((cxc + 1.f) * WLR - 1.f) * 0.5f + 0.5f), 0), WLR - 1);
            float oy = -1.f + (2.f * iy + 1.f) / HLR;
            float ox = -1.f + (2.f * ix + 1.f) / WLR;
            rel0[j] = (cy - oy) * HLR;
            rel1[j] = (cx - ox) * WLR;
            area[j] = fabsf(rel0[j] * rel1[j]) + 1e-9f;
            pix[j]  = iy * WLR + ix;
        }
    }
    float tot = area[0] + area[1] + area[2] + area[3];
    float wgt[4] = { area[3] / tot, area[2] / tot, area[1] / tot, area[0] / tot };
    _Float16* g = grid + (size_t)q * KP0;
    #pragma unroll
    for (int jj = 0; jj < 4; ++jj) {
        g[2 * jj]     = (_Float16)rel0[jj];
        g[2 * jj + 1] = (_Float16)rel1[jj];
    }
    const float twopi = 6.283185307179586f;
    for (int jj = 0; jj < 4; ++jj) {
        float r0 = rel0[jj], r1 = rel1[jj], wj = wgt[jj];
        int pj = pix[jj];
        for (int c = 0; c < 64; ++c) {
            float fs = feat[c * NPIX_LR + pj] * wj;
            float s  = __sinf(twopi * (modw[2 * c] * r0 + modw[2 * c + 1] * r1 + modb[c]));
            g[8   + jj * 64 + c] = (_Float16)fs;
            g[264 + jj * 64 + c] = (_Float16)(fs * s);
        }
    }
    g[520] = (_Float16)(cell[0] * HLR);
    g[521] = (_Float16)(cell[1] * WLR);
    for (int k = K522; k < KP0; ++k) g[k] = (_Float16)0.f;
}

// f32 [N,K] weights -> f16 [N, ldk] (zero-padded K), i.e. WMMA B layout rows.
__global__ void k_cvt_w(const float* __restrict__ src, _Float16* __restrict__ dst,
                        int N, int K, int ldk)
{
    int idx = blockIdx.x * blockDim.x + threadIdx.x;
    if (idx >= N * ldk) return;
    int n = idx / ldk, k = idx % ldk;
    dst[idx] = (k < K) ? (_Float16)src[(size_t)n * K + k] : (_Float16)0.f;
}

__global__ void k_zero(float* __restrict__ p, int n)
{
    int i = blockIdx.x * blockDim.x + threadIdx.x;
    if (i < n) p[i] = 0.f;
}

__device__ __forceinline__ float gelu_exact(float v)
{
    return 0.5f * v * (1.f + erff(v * 0.70710678118654752f));
}

// Fragment loads. brow/arow already include the lane's hi/lo K sub-offset.
__device__ __forceinline__ void load_bfrag(const _Float16* __restrict__ brow,
                                           int k0, HFrag& f)
{
    f.q[0] = *(const v4f*)(brow + k0);
    f.q[1] = *(const v4f*)(brow + k0 + 16);
}

__device__ __forceinline__ void load_afrag(const _Float16* __restrict__ arow,
                                           size_t tstride, int k0, HFrag f[4])
{
    #pragma unroll
    for (int t = 0; t < 4; ++t) {
        const _Float16* ap = arow + t * tstride + k0;
        f[t].q[0] = *(const v4f*)(ap);
        f[t].q[1] = *(const v4f*)(ap + 16);
    }
}

__device__ __forceinline__ void wmma4(const HFrag a[4], const HFrag& b, v8f acc[4])
{
    #pragma unroll
    for (int t = 0; t < 4; ++t)
        acc[t] = __builtin_amdgcn_wmma_f32_16x16x32_f16(
            false, a[t].h, false, b.h, (short)0, acc[t], false, false);
}

// ---------------------------------------------------------------------------
// f16 WMMA GEMM: out[M,N] = A[M,K] * B[N,K]^T + bias (+epilogue).
// One wave owns a 64x16 tile (4 M-subtiles reuse one B fragment).
// K loop is software-pipelined with two register fragment sets (ping-pong):
// loads for step s+1 are issued before the WMMA burst of step s, so the
// s_wait_loadcnt before each WMMA covers loads issued one full step earlier.
// Fragment packing follows the CDNA5 VGPR layouts (ISA 7.12.2):
//   A 16x32 f16 : lane<16 -> K [k0..k0+7],[k0+16..k0+23]; lane>=16 -> +8/+24
//   B 32x16 f16 : lane holds col n0+(lane&15), K same as A ([k0..k0+7],[k0+16..k0+23]; +8 hi)
//   C/D f32     : VGPR r -> row m0+16t+r(+8 hi-half), col n0+(lane&15)
// EPI: 0 = bias, 1 = bias+GELU, 2 = bias+residual add.
// ---------------------------------------------------------------------------
template <int EPI>
__global__ __launch_bounds__(256)
void k_gemm_wmma(const _Float16* __restrict__ A, int lda,
                 const _Float16* __restrict__ Bw, int ldb,
                 const float* __restrict__ bias,
                 const _Float16* __restrict__ resid,
                 _Float16* __restrict__ out, int ldo,
                 int Mrows, int Ncols, int K)
{
    const int wave = blockIdx.x * (blockDim.x >> 5) + (threadIdx.x >> 5);
    const int nT = Ncols >> 4;
    const int mB = Mrows >> 6;
    if (wave >= nT * mB) return;           // wave-uniform: EXEC stays all-ones
    const int m0 = (wave / nT) << 6;
    const int n0 = (wave % nT) << 4;
    const int lane = threadIdx.x & 31;
    const int l15 = lane & 15;
    const bool hi = lane >= 16;

    const _Float16* arow = A  + (size_t)(m0 + l15) * lda + (hi ? 8 : 0);
    const _Float16* brow = Bw + (size_t)(n0 + l15) * ldb + (hi ? 8 : 0);
    const size_t tstride = (size_t)16 * lda;

    const v8f zero8 = {0.f, 0.f, 0.f, 0.f, 0.f, 0.f, 0.f, 0.f};
    v8f acc[4] = {zero8, zero8, zero8, zero8};

    HFrag a0[4], a1[4], b0, b1;
    load_bfrag(brow, 0, b0);
    load_afrag(arow, tstride, 0, a0);

    int k0 = 0;
    while (k0 + 64 <= K) {
        // fetch step s+1 into set1, then run WMMAs on set0
        load_bfrag(brow, k0 + 32, b1);
        load_afrag(arow, tstride, k0 + 32, a1);
        wmma4(a0, b0, acc);
        // fetch step s+2 into set0 (if it exists), then run WMMAs on set1
        if (k0 + 64 < K) {
            load_bfrag(brow, k0 + 64, b0);
            load_afrag(arow, tstride, k0 + 64, a0);
            if (k0 + 128 < K)
                __builtin_prefetch(arow + k0 + 128, 0, 1);   // global_prefetch_b8
        }
        wmma4(a1, b1, acc);
        k0 += 64;
    }
    if (k0 < K)          // odd tail step (K/32 odd): fragments already resident
        wmma4(a0, b0, acc);

    const float bc = bias ? bias[n0 + l15] : 0.f;
    #pragma unroll
    for (int t = 0; t < 4; ++t) {
        #pragma unroll
        for (int r = 0; r < 8; ++r) {
            int row = m0 + t * 16 + r + (hi ? 8 : 0);
            int col = n0 + l15;
            float v = acc[t][r] + bc;
            if (EPI == 1) v = gelu_exact(v);
            if (EPI == 2) v += (float)resid[(size_t)row * ldo + col];
            out[(size_t)row * ldo + col] = (_Float16)v;
        }
    }
}

// ---------------------------------------------------------------------------
// Galerkin attention: layernorm(k), layernorm(v), accumulate kv = k^T v.
// Block = 256 queries x 1 head; normalized k/v staged in LDS, then thread t
// owns output element (i=t/16, j=t%16) and reduces over the 256 rows;
// fp32 global atomics combine the 400 partial blocks.
// ---------------------------------------------------------------------------
__global__ void k_attn_kv(const _Float16* __restrict__ qkv,
                          const float* __restrict__ kg, const float* __restrict__ kb,
                          const float* __restrict__ vg, const float* __restrict__ vb,
                          float* __restrict__ kv)
{
    __shared__ float kbuf[256 * 16];
    __shared__ float vbuf[256 * 16];
    const int tid  = threadIdx.x;
    const int head = blockIdx.y;
    const int n    = blockIdx.x * 256 + tid;
    const _Float16* p = qkv + (size_t)n * 768 + head * 48;   // [q|k|v] x16 per head
    float kx[16], vx[16];
    float km = 0.f, vm = 0.f;
    #pragma unroll
    for (int c = 0; c < 16; ++c) {
        kx[c] = (float)p[16 + c];
        vx[c] = (float)p[32 + c];
        km += kx[c]; vm += vx[c];
    }
    km *= (1.f / 16.f); vm *= (1.f / 16.f);
    float ks2 = 0.f, vs2 = 0.f;
    #pragma unroll
    for (int c = 0; c < 16; ++c) {
        float a = kx[c] - km, b = vx[c] - vm;
        ks2 += a * a; vs2 += b * b;
    }
    float kr = rsqrtf(ks2 * (1.f / 16.f) + 1e-5f);
    float vr = rsqrtf(vs2 * (1.f / 16.f) + 1e-5f);
    #pragma unroll
    for (int c = 0; c < 16; ++c) {
        kbuf[tid * 16 + c] = (kx[c] - km) * kr * kg[head * 16 + c] + kb[head * 16 + c];
        vbuf[tid * 16 + c] = (vx[c] - vm) * vr * vg[head * 16 + c] + vb[head * 16 + c];
    }
    __syncthreads();
    const int i = tid >> 4, jj = tid & 15;
    float s = 0.f;
    for (int nn = 0; nn < 256; ++nn)
        s += kbuf[nn * 16 + i] * vbuf[nn * 16 + jj];
    atomicAdd(&kv[(head * 16 + i) * 16 + jj], s);
}

// o = q @ (kv / N) per head; ret = o + x (attention bias path). kv in LDS.
__global__ void k_attn_apply(const _Float16* __restrict__ qkv,
                             const float* __restrict__ kv,
                             const _Float16* __restrict__ xin,
                             _Float16* __restrict__ ret)
{
    __shared__ float kvs[HEADS * 16 * 16];
    const int tid = threadIdx.x;
    for (int i = tid; i < HEADS * 256; i += 256) kvs[i] = kv[i] * (1.f / (float)MQ);
    __syncthreads();
    const int n = blockIdx.x * 256 + tid;
    const _Float16* qp = qkv + (size_t)n * 768;
    for (int h = 0; h < HEADS; ++h) {
        float q[16];
        #pragma unroll
        for (int c = 0; c < 16; ++c) q[c] = (float)qp[h * 48 + c];
        #pragma unroll
        for (int d = 0; d < 16; ++d) {
            float acc = 0.f;
            #pragma unroll
            for (int c = 0; c < 16; ++c) acc += q[c] * kvs[(h * 16 + c) * 16 + d];
            int col = h * 16 + d;
            ret[(size_t)n * 256 + col] = (_Float16)(acc + (float)xin[(size_t)n * 256 + col]);
        }
    }
}

// fc2 (N=3) fused with bilinear-border skip and final store (fp32 NCHW out).
__global__ void k_final(const _Float16* __restrict__ u,
                        const float* __restrict__ w2, const float* __restrict__ b2,
                        const float* __restrict__ inp, const float* __restrict__ coord,
                        float* __restrict__ out)
{
    int n = blockIdx.x * blockDim.x + threadIdx.x;
    if (n >= MQ) return;
    const _Float16* up = u + (size_t)n * 256;
    float o[3] = {b2[0], b2[1], b2[2]};
    for (int c = 0; c < 256; ++c) {
        float uv = (float)up[c];
        o[0] += uv * w2[c];
        o[1] += uv * w2[256 + c];
        o[2] += uv * w2[512 + c];
    }
    float cy = coord[2 * n], cx = coord[2 * n + 1];
    float iy = fminf(fmaxf(((cy + 1.f) * HLR - 1.f) * 0.5f, 0.f), (float)(HLR - 1));
    float ix = fminf(fmaxf(((cx + 1.f) * WLR - 1.f) * 0.5f, 0.f), (float)(WLR - 1));
    float y0 = floorf(iy), x0 = floorf(ix);
    float wy = iy - y0, wx = ix - x0;
    int y0i = min(max((int)y0, 0), HLR - 1); int y1i = min(y0i + 1, HLR - 1);
    int x0i = min(max((int)x0, 0), WLR - 1); int x1i = min(x0i + 1, WLR - 1);
    #pragma unroll
    for (int ch = 0; ch < 3; ++ch) {
        const float* im = inp + ch * NPIX_LR;
        float v00 = im[y0i * WLR + x0i], v01 = im[y0i * WLR + x1i];
        float v10 = im[y1i * WLR + x0i], v11 = im[y1i * WLR + x1i];
        float bv = (1.f - wy) * ((1.f - wx) * v00 + wx * v01)
                 +        wy  * ((1.f - wx) * v10 + wx * v11);
        out[(size_t)ch * MQ + n] = o[ch] + bv;
    }
}

// ---------------------------------------------------------------------------
// Host-side orchestration.
// ---------------------------------------------------------------------------
extern "C" void kernel_launch(void* const* d_in, const int* in_sizes, int n_in,
                              void* d_out, int out_size, void* d_ws, size_t ws_size,
                              hipStream_t stream)
{
    (void)in_sizes; (void)n_in; (void)out_size; (void)ws_size;

    const float* inp   = (const float*)d_in[0];
    const float* coord = (const float*)d_in[1];
    const float* cell  = (const float*)d_in[2];
    const float* w_in  = (const float*)d_in[3];
    const float* b_in  = (const float*)d_in[4];
    const float* rw1   = (const float*)d_in[5];
    const float* rb1   = (const float*)d_in[6];
    const float* rw2   = (const float*)d_in[7];
    const float* rb2   = (const float*)d_in[8];
    const float* w_out = (const float*)d_in[9];
    const float* b_out = (const float*)d_in[10];
    const float* modw  = (const float*)d_in[11];
    const float* modb  = (const float*)d_in[12];
    const float* c00w  = (const float*)d_in[13];
    const float* c00b  = (const float*)d_in[14];
    const float* fc1w  = (const float*)d_in[35];
    const float* fc1b  = (const float*)d_in[36];
    const float* fc2w  = (const float*)d_in[37];
    const float* fc2b  = (const float*)d_in[38];

    char* ws = (char*)d_ws;
    size_t off = 0;
    auto alloc = [&](size_t bytes) -> char* {
        char* p = ws + off;
        off += (bytes + 255) & ~(size_t)255;
        return p;
    };

    const size_t ENCB = (size_t)64 * NPIX_LR * sizeof(float);   // 1.6 MB
    float* encA = (float*)alloc(ENCB);   // h
    float* encB = (float*)alloc(ENCB);
    float* encC = (float*)alloc(ENCB);
    float* encD = (float*)alloc(ENCB);
    float* feat = (float*)alloc(ENCB);
    float* kvbuf = (float*)alloc((size_t)HEADS * 16 * 16 * sizeof(float));

    _Float16* wc00 = (_Float16*)alloc((size_t)256 * KP0 * 2);
    _Float16* wqkv[2], *wo1[2], *wo2[2];
    for (int p = 0; p < 2; ++p) {
        wqkv[p] = (_Float16*)alloc((size_t)768 * 256 * 2);
        wo1[p]  = (_Float16*)alloc((size_t)256 * 256 * 2);
        wo2[p]  = (_Float16*)alloc((size_t)256 * 256 * 2);
    }
    _Float16* wfc1 = (_Float16*)alloc((size_t)256 * 256 * 2);

    _Float16* gridb = (_Float16*)alloc((size_t)MQ * KP0 * 2);   // 111 MB
    _Float16* xA    = (_Float16*)alloc((size_t)MQ * 256 * 2);   // 52 MB
    _Float16* retb  = (_Float16*)alloc((size_t)MQ * 256 * 2);
    _Float16* hb    = (_Float16*)alloc((size_t)MQ * 256 * 2);
    _Float16* qkvb  = (_Float16*)alloc((size_t)MQ * 768 * 2);   // 157 MB

    const int TB = 256;
    const int gConv = (64 * NPIX_LR + TB - 1) / TB;

    // ---- EDSR encoder ----
    k_conv3x3<<<gConv, TB, 0, stream>>>(inp, 3, w_in, b_in, encA, 64, nullptr, 0);
    const float* bodyIn[4] = {encA, encB, encD, encB};
    float*       bodyOut[4] = {encB, encD, encB, encD};
    for (int i = 0; i < 4; ++i) {
        k_conv3x3<<<gConv, TB, 0, stream>>>(bodyIn[i], 64,
            rw1 + (size_t)i * 64 * 64 * 9, rb1 + (size_t)i * 64, encC, 64, nullptr, 1);
        k_conv3x3<<<gConv, TB, 0, stream>>>(encC, 64,
            rw2 + (size_t)i * 64 * 64 * 9, rb2 + (size_t)i * 64, bodyOut[i], 64, bodyIn[i], 0);
    }
    k_conv3x3<<<gConv, TB, 0, stream>>>(encD, 64, w_out, b_out, feat, 64, encA, 0);

    // ---- grid assembly (A matrix for conv00) ----
    k_build_grid<<<MQ / TB, TB, 0, stream>>>(feat, coord, cell, modw, modb, gridb);

    // ---- weight conversion to f16 B layout ----
    auto cvt = [&](const float* s, _Float16* dptr, int N, int K, int ldk) {
        int n = N * ldk;
        k_cvt_w<<<(n + TB - 1) / TB, TB, 0, stream>>>(s, dptr, N, K, ldk);
    };
    cvt(c00w, wc00, 256, K522, KP0);
    for (int p = 0; p < 2; ++p) {
        cvt((const float*)d_in[15 + 10 * p + 0], wqkv[p], 768, 256, 256);
        cvt((const float*)d_in[15 + 10 * p + 6], wo1[p], 256, 256, 256);
        cvt((const float*)d_in[15 + 10 * p + 8], wo2[p], 256, 256, 256);
    }
    cvt(fc1w, wfc1, 256, 256, 256);

    auto gemm_blocks = [](int M, int N) { return ((M / 64) * (N / 16) + 7) / 8; };

    // ---- conv00: grid[MQ,544] x W[256,544]^T -> x ----
    k_gemm_wmma<0><<<gemm_blocks(MQ, 256), TB, 0, stream>>>(
        gridb, KP0, wc00, KP0, c00b, nullptr, xA, 256, MQ, 256, KP0);

    // ---- two Galerkin attention blocks ----
    for (int p = 0; p < 2; ++p) {
        const float* qkvB = (const float*)d_in[15 + 10 * p + 1];
        const float* klg  = (const float*)d_in[15 + 10 * p + 2];
        const float* klb  = (const float*)d_in[15 + 10 * p + 3];
        const float* vlg  = (const float*)d_in[15 + 10 * p + 4];
        const float* vlb  = (const float*)d_in[15 + 10 * p + 5];
        const float* o1b  = (const float*)d_in[15 + 10 * p + 7];
        const float* o2b  = (const float*)d_in[15 + 10 * p + 9];

        k_gemm_wmma<0><<<gemm_blocks(MQ, 768), TB, 0, stream>>>(
            xA, 256, wqkv[p], 256, qkvB, nullptr, qkvb, 768, MQ, 768, 256);

        k_zero<<<(HEADS * 256 + TB - 1) / TB, TB, 0, stream>>>(kvbuf, HEADS * 256);
        dim3 gkv(MQ / 256, HEADS);
        k_attn_kv<<<gkv, TB, 0, stream>>>(qkvb, klg, klb, vlg, vlb, kvbuf);
        k_attn_apply<<<MQ / 256, TB, 0, stream>>>(qkvb, kvbuf, xA, retb);

        k_gemm_wmma<1><<<gemm_blocks(MQ, 256), TB, 0, stream>>>(
            retb, 256, wo1[p], 256, o1b, nullptr, hb, 256, MQ, 256, 256);
        k_gemm_wmma<2><<<gemm_blocks(MQ, 256), TB, 0, stream>>>(
            hb, 256, wo2[p], 256, o2b, xA /*residual = block input*/, xA, 256, MQ, 256, 256);
    }

    // ---- fc1 (GELU) + fused fc2/bilinear output ----
    k_gemm_wmma<1><<<gemm_blocks(MQ, 256), TB, 0, stream>>>(
        xA, 256, wfc1, 256, fc1b, nullptr, hb, 256, MQ, 256, 256);
    k_final<<<MQ / TB, TB, 0, stream>>>(hb, fc2w, fc2b, inp, coord, (float*)d_out);
}
